// SyndromeTransformerLayer_73100343378493
// MI455X (gfx1250) — matrix operand order repaired and checked by reading, not compiled, not run on hardware
//
#include <hip/hip_runtime.h>
#include <hip/hip_bf16.h>

typedef unsigned int uint;
typedef __attribute__((ext_vector_type(16))) _Float16 v16h;
typedef __attribute__((ext_vector_type(8)))  float    v8f;

#define LDSA 40   // LDS row pitch in halves (80 B = 16B-aligned, pad vs bank conflicts)

// ---------------------------------------------------------------------------
// WMMA fragment loaders (CDNA5 16-bit layouts, wave32)
// A 16x32 f16: lane 0-15 = M, VGPR v: K = 16*(v>=4) + 8*(lane>=16) + 2*(v&3)
// B 32x16 f16: lane%16 = N, VGPR v: K = 16*(v>=4) + 8*(lane>=16) + 2*(v&3)
// ---------------------------------------------------------------------------
__device__ __forceinline__ v16h frag_a(const _Float16* As, int m, int hlf) {
  union { uint u[8]; v16h v; } t;
#pragma unroll
  for (int i = 0; i < 8; ++i) {
    int k = 16 * (i >> 2) + 8 * hlf + 2 * (i & 3);
    t.u[i] = *(const uint*)(As + m * LDSA + k);
  }
  return t.v;
}
__device__ __forceinline__ v16h frag_b(const _Float16* Bs, int n, int hlf) {
  union { uint u[8]; v16h v; } t;
#pragma unroll
  for (int i = 0; i < 8; ++i) {
    int k = 16 * (i >> 2) + 8 * hlf + 2 * (i & 3);
    t.u[i] = *(const uint*)(Bs + n * LDSA + k);
  }
  return t.v;
}
__device__ __forceinline__ v8f wmma32(v16h a, v16h b, v8f c) {
  return __builtin_amdgcn_wmma_f32_16x16x32_f16(false, a, false, b, (short)0, c, false, false);
}

// Async 16-byte global->LDS copy (CDNA5 GLOBAL_LOAD_ASYNC_TO_LDS_B128, ASYNCcnt)
__device__ __forceinline__ void async_b128(_Float16* lds, const _Float16* gp) {
  unsigned loff = (unsigned)(size_t)lds;
  asm volatile("global_load_async_to_lds_b128 %0, %1, off"
               :: "v"(loff), "v"(gp) : "memory");
}
__device__ __forceinline__ void wait_async() {
  asm volatile("s_wait_asynccnt 0x0" ::: "memory");
}

// ---------------------------------------------------------------------------
// f32 -> f16 convert (grid-stride)
// ---------------------------------------------------------------------------
__global__ void cvt_f16_kernel(const float* __restrict__ s, _Float16* __restrict__ d, long n) {
  long i  = (long)blockIdx.x * blockDim.x + threadIdx.x;
  long st = (long)gridDim.x * blockDim.x;
  for (; i < n; i += st) d[i] = (_Float16)s[i];
}

// ---------------------------------------------------------------------------
// LayerNorm over D=512, one row per block (256 thr, 2 elems each).
// out = LN(x + res) * g + b ; writes fp32 and/or f16.
// ---------------------------------------------------------------------------
__global__ __launch_bounds__(256) void ln_kernel(
    const float* __restrict__ x, const float* __restrict__ res,
    const float* __restrict__ g, const float* __restrict__ bb,
    float* __restrict__ outF, _Float16* __restrict__ outH) {
  long row = blockIdx.x;
  const float* xr = x + row * 512;
  const float* rr = res ? res + row * 512 : nullptr;
  int tid = threadIdx.x;
  float v0 = xr[tid]       + (rr ? rr[tid]       : 0.f);
  float v1 = xr[tid + 256] + (rr ? rr[tid + 256] : 0.f);
  __shared__ float red[256];
  red[tid] = v0 + v1; __syncthreads();
  for (int s = 128; s > 0; s >>= 1) { if (tid < s) red[tid] += red[tid + s]; __syncthreads(); }
  float mean = red[0] * (1.f / 512.f);
  __syncthreads();
  float d0 = v0 - mean, d1 = v1 - mean;
  red[tid] = d0 * d0 + d1 * d1; __syncthreads();
  for (int s = 128; s > 0; s >>= 1) { if (tid < s) red[tid] += red[tid + s]; __syncthreads(); }
  float inv = rsqrtf(red[0] * (1.f / 512.f) + 1e-5f);
  float y0 = d0 * inv * g[tid] + bb[tid];
  float y1 = d1 * inv * g[tid + 256] + bb[tid + 256];
  if (outF) { outF[row * 512 + tid] = y0; outF[row * 512 + tid + 256] = y1; }
  if (outH) { outH[row * 512 + tid] = (_Float16)y0; outH[row * 512 + tid + 256] = (_Float16)y1; }
}

// ---------------------------------------------------------------------------
// Dilated-conv stack as implicit GEMM, async-staged.
// Block = 256 thr (8 waves, 2x4), tile = 64 tokens x 128 out-ch,
// wave tile 32x32 (4 WMMA accumulators). K loop: 27 taps x 512 in-ch.
// Zero-padded gather: valid lanes use GLOBAL_LOAD_ASYNC_TO_LDS_B128,
// out-of-grid lanes store zeros to LDS.
// ---------------------------------------------------------------------------
__global__ __launch_bounds__(256) void conv_wmma_kernel(
    const _Float16* __restrict__ x16, const _Float16* __restrict__ w16,
    const float* __restrict__ conv_b, float* __restrict__ out) {
  __shared__ _Float16 As[64 * LDSA];
  __shared__ _Float16 Bs[128 * LDSA];
  const int S_ = 576, D_ = 512, G = 24;
  int tid = threadIdx.x, lane = tid & 31, wave = tid >> 5;
  int wm = wave >> 2, wn = wave & 3;
  int m0 = blockIdx.x * 64, n0 = blockIdx.y * 128;
  int hlf = lane >> 4, l15 = lane & 15;
  v8f acc00 = {0.f,0.f,0.f,0.f,0.f,0.f,0.f,0.f}, acc01 = acc00, acc10 = acc00, acc11 = acc00;

  // this thread's fixed A slot: 64 rows x 4 chunks of 16B
  int arow = tid >> 2, ach = tid & 3;
  int am = m0 + arow, as = am % S_;
  int abase = am - as;            // batch*S
  int ar = as / G, ac = as % G;

  for (int t = 0; t < 27; ++t) {
    int dil = 1 << (t / 9);
    int dy = ((t % 9) / 3 - 1) * dil, dx = (t % 3 - 1) * dil;
    long wbase = (long)t * D_ * D_;
    int rr2 = ar + dy, cc = ac + dx;
    bool valid = (rr2 >= 0 && rr2 < G && cc >= 0 && cc < G);
    long asrc = (long)(abase + rr2 * G + cc) * D_;
    __builtin_prefetch(w16 + wbase + n0, 0, 1);
    for (int k0 = 0; k0 < D_; k0 += 32) {
      // A: async DMA (or zero-fill) of 64 x 32 halves
      if (valid) {
        async_b128(As + arow * LDSA + ach * 8, x16 + asrc + k0 + ach * 8);
      } else {
        uint4 zz = {0, 0, 0, 0};
        *(uint4*)(As + arow * LDSA + ach * 8) = zz;
      }
      // B: w[tap][k][n] -> Bs[n][k] (transposed for contiguous k-pairs)
#pragma unroll
      for (int r = 0; r < 8; ++r) {
        int idx = tid + r * 256, k = idx >> 6, np = idx & 63;
        uint val = *(const uint*)(w16 + wbase + (long)(k0 + k) * D_ + n0 + 2 * np);
        _Float16* p = (_Float16*)&val;
        Bs[(2 * np)     * LDSA + k] = p[0];
        Bs[(2 * np + 1) * LDSA + k] = p[1];
      }
      wait_async();
      __syncthreads();
      v16h a0 = frag_a(As, 32 * wm + l15, hlf);
      v16h a1 = frag_a(As, 32 * wm + 16 + l15, hlf);
      v16h b0 = frag_b(Bs, 32 * wn + l15, hlf);
      v16h b1 = frag_b(Bs, 32 * wn + 16 + l15, hlf);
      acc00 = wmma32(a0, b0, acc00);
      acc01 = wmma32(a0, b1, acc01);
      acc10 = wmma32(a1, b0, acc10);
      acc11 = wmma32(a1, b1, acc11);
      __syncthreads();
    }
  }
  v8f accs[2][2] = {{acc00, acc01}, {acc10, acc11}};
#pragma unroll
  for (int mf = 0; mf < 2; ++mf) {
#pragma unroll
    for (int nf = 0; nf < 2; ++nf) {
      int n = n0 + 32 * wn + 16 * nf + l15;
      float cb = conv_b[n] + conv_b[512 + n] + conv_b[1024 + n];
      int mb = m0 + 32 * wm + 16 * mf + (lane >> 4) * 8;
#pragma unroll
      for (int r2 = 0; r2 < 8; ++r2)
        out[(long)(mb + r2) * 512 + n] = accs[mf][nf][r2] + cb;
    }
  }
}

// ---------------------------------------------------------------------------
// Static pair-bias: tables -> 48 -> relu MLP -> 48 -> project to H=8 heads.
// ---------------------------------------------------------------------------
__global__ __launch_bounds__(256) void static_bias_kernel(
    const int* __restrict__ idx,
    const float* __restrict__ row_w, const float* __restrict__ col_w,
    const float* __restrict__ dx_w, const float* __restrict__ dy_w,
    const float* __restrict__ manh_w, const float* __restrict__ same_w,
    const float* __restrict__ mlp_w1, const float* __restrict__ mlp_b1,
    const float* __restrict__ mlp_w2, const float* __restrict__ mlp_b2,
    const float* __restrict__ bias_w, const float* __restrict__ bias_b,
    float* __restrict__ sbias) {
  __shared__ float W1[2304], W2[2304], BW[384], B1[48], B2[48], BB[8];
  int tid = threadIdx.x;
  for (int i = tid; i < 2304; i += 256) { W1[i] = mlp_w1[i]; W2[i] = mlp_w2[i]; }
  for (int i = tid; i < 384; i += 256) BW[i] = bias_w[i];
  if (tid < 48) { B1[tid] = mlp_b1[tid]; B2[tid] = mlp_b2[tid]; }
  if (tid < 8) BB[tid] = bias_b[tid];
  __syncthreads();
  long p = (long)blockIdx.x * 256 + tid;
  if (p >= 576L * 576L) return;
  const int* ip = idx + p * 8;
  const float* tabs[8] = {row_w, col_w, row_w, col_w, dx_w, dy_w, manh_w, same_w};
  float e[48];
#pragma unroll
  for (int t = 0; t < 8; ++t) {
    const float* src = tabs[t] + ip[t] * 6;
#pragma unroll
    for (int j = 0; j < 6; ++j) e[t * 6 + j] = src[j];
  }
  float h1[48];
  for (int o = 0; o < 48; ++o) {
    float s = B1[o];
    for (int c = 0; c < 48; ++c) s += e[c] * W1[c * 48 + o];
    h1[o] = fmaxf(s, 0.f);
  }
  float h2[48];
  for (int o = 0; o < 48; ++o) {
    float s = B2[o];
    for (int c = 0; c < 48; ++c) s += h1[c] * W2[c * 48 + o];
    h2[o] = s;
  }
#pragma unroll
  for (int h = 0; h < 8; ++h) {
    float s = BB[h];
    for (int c = 0; c < 48; ++c) s += h2[c] * BW[c * 8 + h];
    sbias[p * 8 + h] = s;
  }
}

// ---------------------------------------------------------------------------
// High-intensity WMMA GEMM: tile 64(M)x128(N), 8 waves (2x4), wave tile 32x32
// (4 accumulators). A staged via async DMA; B staged with transpose.
// Requires M%64==0, N%128==0, K%32==0.
// ---------------------------------------------------------------------------
__global__ __launch_bounds__(256) void gemm_big_kernel(
    const _Float16* __restrict__ A, const _Float16* __restrict__ Bm,
    float* __restrict__ Cf, _Float16* __restrict__ Ch,
    const float* __restrict__ bias,
    int M, int N, int K,
    long aRow, long bK, long bN, long cRow) {
  __shared__ _Float16 As[64 * LDSA];
  __shared__ _Float16 Bs[128 * LDSA];
  int tid = threadIdx.x, lane = tid & 31, wave = tid >> 5;
  int wm = wave >> 2, wn = wave & 3;
  int m0 = blockIdx.x * 64, n0 = blockIdx.y * 128;
  int hlf = lane >> 4, l15 = lane & 15;
  v8f acc00 = {0.f,0.f,0.f,0.f,0.f,0.f,0.f,0.f}, acc01 = acc00, acc10 = acc00, acc11 = acc00;
  int arow = tid >> 2, ach = tid & 3;
  const _Float16* aRowPtr = A + (long)(m0 + arow) * aRow + ach * 8;

  for (int k0 = 0; k0 < K; k0 += 32) {
    // A: 64 rows x 32 halves via async DMA (one b128 per thread)
    async_b128(As + arow * LDSA + ach * 8, aRowPtr + k0);
    // B: 32k x 128n -> Bs[n][k]
    if (bN == 1) {
#pragma unroll
      for (int r = 0; r < 8; ++r) {
        int idx = tid + r * 256, k = idx >> 6, np = idx & 63;
        uint val = *(const uint*)(Bm + (long)(k0 + k) * bK + n0 + 2 * np);
        _Float16* p = (_Float16*)&val;
        Bs[(2 * np)     * LDSA + k] = p[0];
        Bs[(2 * np + 1) * LDSA + k] = p[1];
      }
    } else {
#pragma unroll
      for (int r = 0; r < 16; ++r) {
        int idx = tid + r * 256, k = idx >> 7, n = idx & 127;
        Bs[n * LDSA + k] = Bm[(long)(k0 + k) * bK + (long)(n0 + n) * bN];
      }
    }
    if (k0 + 32 < K) __builtin_prefetch(Bm + (long)(k0 + 32) * bK + n0, 0, 1);
    wait_async();
    __syncthreads();
    v16h a0 = frag_a(As, 32 * wm + l15, hlf);
    v16h a1 = frag_a(As, 32 * wm + 16 + l15, hlf);
    v16h b0 = frag_b(Bs, 32 * wn + l15, hlf);
    v16h b1 = frag_b(Bs, 32 * wn + 16 + l15, hlf);
    acc00 = wmma32(a0, b0, acc00);
    acc01 = wmma32(a0, b1, acc01);
    acc10 = wmma32(a1, b0, acc10);
    acc11 = wmma32(a1, b1, acc11);
    __syncthreads();
  }
  v8f accs[2][2] = {{acc00, acc01}, {acc10, acc11}};
#pragma unroll
  for (int mf = 0; mf < 2; ++mf) {
#pragma unroll
    for (int nf = 0; nf < 2; ++nf) {
      int n = n0 + 32 * wn + 16 * nf + l15;
      float bv = bias ? bias[n] : 0.f;
      int mb = m0 + 32 * wm + 16 * mf + (lane >> 4) * 8;
#pragma unroll
      for (int r2 = 0; r2 < 8; ++r2) {
        float val = accs[mf][nf][r2] + bv;
        if (Cf) Cf[(long)(mb + r2) * cRow + n] = val;
        if (Ch) Ch[(long)(mb + r2) * cRow + n] = (_Float16)val;
      }
    }
  }
}

// ---------------------------------------------------------------------------
// Small strided/batched WMMA GEMM (tile 32x64) — used for attn @ v (N=64).
// ---------------------------------------------------------------------------
__global__ __launch_bounds__(256) void gemm_wmma_kernel(
    const _Float16* __restrict__ A, const _Float16* __restrict__ Bm,
    float* __restrict__ Cf, _Float16* __restrict__ Ch,
    const float* __restrict__ bias,
    int M, int N, int K,
    long aRow, long aS1, long aS2,
    long bK, long bN, long bS1, long bS2,
    long cRow, long cS1, long cS2, int zDiv) {
  __shared__ _Float16 As[32 * LDSA];
  __shared__ _Float16 Bs[64 * LDSA];
  int tid = threadIdx.x, lane = tid & 31, wave = tid >> 5;
  int wm = wave >> 2, wn = wave & 3;
  int z = blockIdx.z, z1 = z / zDiv, z2 = z % zDiv;
  long aOff = z1 * aS1 + z2 * aS2;
  long bOff = z1 * bS1 + z2 * bS2;
  long cOff = z1 * cS1 + z2 * cS2;
  int m0 = blockIdx.x * 32, n0 = blockIdx.y * 64;
  int hlf = lane >> 4;
  v8f acc = {0.f, 0.f, 0.f, 0.f, 0.f, 0.f, 0.f, 0.f};
  for (int k0 = 0; k0 < K; k0 += 32) {
#pragma unroll
    for (int r = 0; r < 2; ++r) {           // A: 32 rows x 16 u32
      int idx = tid + r * 256, row = idx >> 4, cp = idx & 15;
      *(uint*)(As + row * LDSA + 2 * cp) =
          *(const uint*)(A + aOff + (long)(m0 + row) * aRow + k0 + 2 * cp);
    }
#pragma unroll
    for (int r = 0; r < 8; ++r) {           // B: 32k x 64n element-wise, strided
      int idx = tid + r * 256, k = idx >> 6, n = idx & 63;
      Bs[n * LDSA + k] = Bm[bOff + (long)(k0 + k) * bK + (long)(n0 + n) * bN];
    }
    __syncthreads();
    v16h af = frag_a(As, 16 * wm + (lane & 15), hlf);
    v16h bf = frag_b(Bs, 16 * wn + (lane & 15), hlf);
    acc = wmma32(af, bf, acc);
    __syncthreads();
  }
  int n = n0 + 16 * wn + (lane & 15);
  int mBase = m0 + 16 * wm + (lane >> 4) * 8;
  float bv = bias ? bias[n] : 0.f;
#pragma unroll
  for (int r2 = 0; r2 < 8; ++r2) {
    int m = mBase + r2;
    float val = acc[r2] + bv;
    if (Cf) Cf[cOff + (long)m * cRow + n] = val;
    if (Ch) Ch[cOff + (long)m * cRow + n] = (_Float16)val;
  }
}

// ---------------------------------------------------------------------------
// Scores = (q . k)/8 + static_bias + dynamic_bias + bias_b, batched over b*H.
// ---------------------------------------------------------------------------
__global__ __launch_bounds__(256) void attn_scores_kernel(
    const _Float16* __restrict__ qkv, const float* __restrict__ sbias,
    const float* __restrict__ events, const float* __restrict__ prev,
    const float* __restrict__ bias_w, const float* __restrict__ bias_b,
    float* __restrict__ scores) {
  __shared__ _Float16 As[32 * LDSA];
  __shared__ _Float16 Bs[64 * LDSA];
  const int S_ = 576;
  int tid = threadIdx.x, lane = tid & 31, wave = tid >> 5;
  int wm = wave >> 2, wn = wave & 3;
  int z = blockIdx.z, b = z >> 3, h = z & 7;
  long qOff = (long)b * S_ * 1536 + h * 64;
  long kOff = qOff + 512;
  int i0 = blockIdx.x * 32, j0 = blockIdx.y * 64;
  int hlf = lane >> 4;
  v8f acc = {0.f, 0.f, 0.f, 0.f, 0.f, 0.f, 0.f, 0.f};
  for (int k0 = 0; k0 < 64; k0 += 32) {
#pragma unroll
    for (int r = 0; r < 2; ++r) {           // q tile
      int idx = tid + r * 256, row = idx >> 4, cp = idx & 15;
      *(uint*)(As + row * LDSA + 2 * cp) =
          *(const uint*)(qkv + qOff + (long)(i0 + row) * 1536 + k0 + 2 * cp);
    }
#pragma unroll
    for (int r = 0; r < 4; ++r) {           // k tile (d contiguous -> pair loads)
      int idx = tid + r * 256, n = idx >> 4, cp = idx & 15;
      *(uint*)(Bs + n * LDSA + 2 * cp) =
          *(const uint*)(qkv + kOff + (long)(j0 + n) * 1536 + k0 + 2 * cp);
    }
    __syncthreads();
    v16h af = frag_a(As, 16 * wm + (lane & 15), hlf);
    v16h bf = frag_b(Bs, 16 * wn + (lane & 15), hlf);
    acc = wmma32(af, bf, acc);
    __syncthreads();
  }
  int j = j0 + 16 * wn + (lane & 15);
  int iBase = i0 + 16 * wm + (lane >> 4) * 8;
  float w48 = bias_w[48 * 8 + h], w49 = bias_w[49 * 8 + h], w50 = bias_w[50 * 8 + h];
  float w51 = bias_w[51 * 8 + h], w52 = bias_w[52 * 8 + h], w53 = bias_w[53 * 8 + h];
  float w54 = bias_w[54 * 8 + h];
  float ej = events[b * S_ + j], pj = prev[b * S_ + j];
  long zOff = (long)z * S_ * S_;
#pragma unroll
  for (int r2 = 0; r2 < 8; ++r2) {
    int i = iBase + r2;
    float ei = events[b * S_ + i], pi = prev[b * S_ + i];
    float dyn = ei * ej * w48 + ei * pj * w49 + pi * ej * w50 + pi * pj * w51;
    if (i == j) dyn += ei * w52 + pi * w53 + ei * pi * w54;
    float val = acc[r2] * 0.125f + sbias[((long)i * S_ + j) * 8 + h] + bias_b[h] + dyn;
    scores[zOff + (long)i * S_ + j] = val;
  }
}

// ---------------------------------------------------------------------------
// Row softmax over 576, in-place fp32 -> f16 (halves packed at row start).
// ---------------------------------------------------------------------------
__global__ __launch_bounds__(256) void softmax_rows_kernel(float* __restrict__ scores) {
  long row = blockIdx.x;
  float* p = scores + row * 576;
  int tid = threadIdx.x;
  float v0 = p[tid], v1 = p[tid + 256];
  float v2 = (tid < 64) ? p[tid + 512] : -3.4e38f;
  __shared__ float red[256];
  red[tid] = fmaxf(fmaxf(v0, v1), v2); __syncthreads();
  for (int s = 128; s > 0; s >>= 1) { if (tid < s) red[tid] = fmaxf(red[tid], red[tid + s]); __syncthreads(); }
  float m = red[0];
  __syncthreads();
  float e0 = __expf(v0 - m), e1 = __expf(v1 - m);
  float e2 = (tid < 64) ? __expf(v2 - m) : 0.f;
  red[tid] = e0 + e1 + e2; __syncthreads();
  for (int s = 128; s > 0; s >>= 1) { if (tid < s) red[tid] += red[tid + s]; __syncthreads(); }
  float inv = 1.f / red[0];
  __syncthreads();
  _Float16* ph = (_Float16*)p;
  ph[tid] = (_Float16)(e0 * inv);
  ph[tid + 256] = (_Float16)(e1 * inv);
  if (tid < 64) ph[tid + 512] = (_Float16)(e2 * inv);
}

// ---------------------------------------------------------------------------
// SwiGLU-style gate: out16 = p * sigmoid(g)
// ---------------------------------------------------------------------------
__global__ void gated_kernel(const float* __restrict__ p, const float* __restrict__ g,
                             _Float16* __restrict__ out, long n) {
  long i  = (long)blockIdx.x * blockDim.x + threadIdx.x;
  long st = (long)gridDim.x * blockDim.x;
  for (; i < n; i += st)
    out[i] = (_Float16)(p[i] * (1.f / (1.f + __expf(-g[i]))));
}

// ---------------------------------------------------------------------------
extern "C" void kernel_launch(void* const* d_in, const int* in_sizes, int n_in,
                              void* d_out, int out_size, void* d_ws, size_t ws_size,
                              hipStream_t stream) {
  (void)in_sizes; (void)n_in; (void)out_size; (void)ws_size;
  const long B = 4, S = 576, D = 512, H = 8, BS = B * S;

  const float* state   = (const float*)d_in[0];
  const float* events  = (const float*)d_in[1];
  const float* prev    = (const float*)d_in[2];
  const float* conv_w  = (const float*)d_in[3];
  const float* conv_b  = (const float*)d_in[4];
  const float* row_w   = (const float*)d_in[5];
  const float* col_w   = (const float*)d_in[6];
  const float* dx_w    = (const float*)d_in[7];
  const float* dy_w    = (const float*)d_in[8];
  const float* manh_w  = (const float*)d_in[9];
  const float* same_w  = (const float*)d_in[10];
  const float* mlp_w1  = (const float*)d_in[11];
  const float* mlp_b1  = (const float*)d_in[12];
  const float* mlp_w2  = (const float*)d_in[13];
  const float* mlp_b2  = (const float*)d_in[14];
  const float* qkv_w   = (const float*)d_in[15];
  const float* qkv_b   = (const float*)d_in[16];
  const float* o_w     = (const float*)d_in[17];
  const float* o_b     = (const float*)d_in[18];
  const float* bias_w  = (const float*)d_in[19];
  const float* bias_b  = (const float*)d_in[20];
  const float* ffp_w   = (const float*)d_in[21];
  const float* ffp_b   = (const float*)d_in[22];
  const float* ffg_w   = (const float*)d_in[23];
  const float* ffg_b   = (const float*)d_in[24];
  const float* ffo_w   = (const float*)d_in[25];
  const float* ffo_b   = (const float*)d_in[26];
  const float* n1_g = (const float*)d_in[27], *n1_b = (const float*)d_in[28];
  const float* n2_g = (const float*)d_in[29], *n2_b = (const float*)d_in[30];
  const float* n3_g = (const float*)d_in[31], *n3_b = (const float*)d_in[32];
  const int*   sidx = (const int*)d_in[33];

  char* ws = (char*)d_ws;
  size_t cur = 0;
  auto take = [&](size_t bytes) { size_t o = cur; cur += (bytes + 255) & ~(size_t)255; return o; };
  size_t off_xlnF    = take(BS * D * 4);
  size_t off_xlnH    = take(BS * D * 2);
  size_t off_convw   = take(27L * D * D * 2);
  size_t off_convout = take(BS * D * 4);
  size_t off_state2F = take(BS * D * 4);       // contiguous with convout (gated alias)
  size_t off_state2H = take(BS * D * 2);
  size_t off_qkvw    = take(D * 3 * D * 2);
  size_t off_qkv16   = take(BS * 3 * D * 2);
  size_t off_sbias   = take(S * S * H * 4);
  size_t off_scores  = take(B * H * S * S * 4);
  size_t off_ao16    = take(BS * D * 2);
  size_t off_ow      = take(D * D * 2);
  size_t off_oproj   = take(BS * D * 4);
  size_t off_state3F = take(BS * D * 4);
  size_t off_state3H = take(BS * D * 2);
  size_t off_ffpw    = take(D * 4 * D * 2);
  size_t off_ffgw    = take(D * 4 * D * 2);
  size_t off_ffow    = take(4 * D * D * 2);
  // aliases (phase-disjoint reuse of large regions)
  size_t off_ffpO  = off_scores;                          // 18.9 MB
  size_t off_ffgO  = off_scores + (size_t)BS * 4 * D * 4; // next 18.9 MB (fits in 42.5)
  size_t off_gated = off_convout;                         // convout+state2F = 9.44 MB
  size_t off_ffoO  = off_qkv16;                           // 4.7 MB into 7.1 MB

  auto F  = [&](size_t o) { return (float*)(ws + o); };
  auto Hp = [&](size_t o) { return (_Float16*)(ws + o); };

  // 1) weights -> f16
  cvt_f16_kernel<<<1024, 256, 0, stream>>>(conv_w, Hp(off_convw), 27L * D * D);
  cvt_f16_kernel<<<512, 256, 0, stream>>>(qkv_w, Hp(off_qkvw), D * 3 * D);
  cvt_f16_kernel<<<256, 256, 0, stream>>>(o_w, Hp(off_ow), D * D);
  cvt_f16_kernel<<<512, 256, 0, stream>>>(ffp_w, Hp(off_ffpw), D * 4 * D);
  cvt_f16_kernel<<<512, 256, 0, stream>>>(ffg_w, Hp(off_ffgw), D * 4 * D);
  cvt_f16_kernel<<<512, 256, 0, stream>>>(ffo_w, Hp(off_ffow), 4 * D * D);

  // 2) LN1
  ln_kernel<<<BS, 256, 0, stream>>>(state, nullptr, n1_g, n1_b, F(off_xlnF), Hp(off_xlnH));

  // 3) dilated conv stack (implicit GEMM, async + WMMA)
  conv_wmma_kernel<<<dim3(BS / 64, D / 128), 256, 0, stream>>>(
      Hp(off_xlnH), Hp(off_convw), conv_b, F(off_convout));

  // 4) state2 = LN(x_ln + conv_out)
  ln_kernel<<<BS, 256, 0, stream>>>(F(off_convout), F(off_xlnF), n3_g, n3_b,
                                    F(off_state2F), Hp(off_state2H));

  // 5) static pair bias projected to heads
  static_bias_kernel<<<(S * S) / 256, 256, 0, stream>>>(
      sidx, row_w, col_w, dx_w, dy_w, manh_w, same_w,
      mlp_w1, mlp_b1, mlp_w2, mlp_b2, bias_w, bias_b, F(off_sbias));

  // 6) QKV GEMM (f16 out)
  gemm_big_kernel<<<dim3(BS / 64, (3 * D) / 128, 1), 256, 0, stream>>>(
      Hp(off_state2H), Hp(off_qkvw), nullptr, Hp(off_qkv16), qkv_b,
      BS, 3 * D, D, 512, 1536, 1, 1536);

  // 7) scores + bias (batched over b*H)
  attn_scores_kernel<<<dim3(S / 32, S / 64, B * H), 256, 0, stream>>>(
      Hp(off_qkv16), F(off_sbias), events, prev, bias_w, bias_b, F(off_scores));

  // 8) softmax (in-place f32 -> f16)
  softmax_rows_kernel<<<B * H * S, 256, 0, stream>>>(F(off_scores));

  // 9) attn @ v (batched), f16 out in (B,S,H,hd) packing
  gemm_wmma_kernel<<<dim3(S / 32, 1, B * H), 256, 0, stream>>>(
      Hp(off_scores), Hp(off_qkv16) + 1024, nullptr, Hp(off_ao16), nullptr,
      S, 64, S,
      /*a*/ 2 * S, H * 2 * S * S, 2 * S * S,
      /*b*/ 1536, 1, S * 1536, 64,
      /*c*/ 512, S * 512, 64, (int)H);

  // 10) O projection
  gemm_big_kernel<<<dim3(BS / 64, D / 128, 1), 256, 0, stream>>>(
      Hp(off_ao16), Hp(off_ow), F(off_oproj), nullptr, o_b,
      BS, D, D, 512, 512, 1, 512);

  // 11) state3 = LN(state2 + attn_out)
  ln_kernel<<<BS, 256, 0, stream>>>(F(off_oproj), F(off_state2F), n2_g, n2_b,
                                    F(off_state3F), Hp(off_state3H));

  // 12) FF up + gate GEMMs
  gemm_big_kernel<<<dim3(BS / 64, (4 * D) / 128, 1), 256, 0, stream>>>(
      Hp(off_state3H), Hp(off_ffpw), F(off_ffpO), nullptr, ffp_b,
      BS, 4 * D, D, 512, 2048, 1, 2048);
  gemm_big_kernel<<<dim3(BS / 64, (4 * D) / 128, 1), 256, 0, stream>>>(
      Hp(off_state3H), Hp(off_ffgw), F(off_ffgO), nullptr, ffg_b,
      BS, 4 * D, D, 512, 2048, 1, 2048);

  // 13) gate
  gated_kernel<<<2048, 256, 0, stream>>>(F(off_ffpO), F(off_ffgO), Hp(off_gated), BS * 4 * D);

  // 14) FF down
  gemm_big_kernel<<<dim3(BS / 64, D / 128, 1), 256, 0, stream>>>(
      Hp(off_gated), Hp(off_ffow), F(off_ffoO), nullptr, ffo_b,
      BS, D, 4 * D, 2048, 512, 1, 512);

  // 15) output = LN(state3 + ff)
  ln_kernel<<<BS, 256, 0, stream>>>(F(off_ffoO), F(off_state3F), n3_g, n3_b,
                                    (float*)d_out, nullptr);
}
